// Decoder_60765197304409
// MI455X (gfx1250) — matrix operand order repaired and checked by reading, not compiled, not run on hardware
//
#include <hip/hip_runtime.h>
#include <hip/hip_bf16.h>

typedef __attribute__((ext_vector_type(16))) _Float16 v16h;
typedef __attribute__((ext_vector_type(8)))  _Float16 v8h;
typedef __attribute__((ext_vector_type(8)))  float    v8f;
typedef __attribute__((ext_vector_type(4)))  float    fv4;

#define VV   30
#define DHD  512
#define EE   256
#define KVD  128
#define BB   256
#define TT   800
#define LL   300
#define K1D  896   // E + KV + DH  (lstm1 input width: [emb | ctx | h1])
#define K2D  640   // DH + KV      (lstm2 input width: [h1 | h2])

__device__ __forceinline__ float sigmoidf_(float x) {
    return 1.0f / (1.0f + __expf(-x));
}

// ---- WMMA tile loaders (16x16x32 f16, wave32; A and B both use contiguous k = 16*(lane/16) + s) ----
__device__ __forceinline__ v16h load_a_tile(const _Float16* __restrict__ A, int ld,
                                            int m0, int k0, int lane) {
    int row = m0 + (lane & 15);
    int kh  = k0 + ((lane >> 4) << 4);
    const _Float16* p = A + (size_t)row * ld + kh;
    v8h lo = *(const v8h*)p;
    v8h hi = *(const v8h*)(p + 8);
    v16h a;
#pragma unroll
    for (int i = 0; i < 8; ++i) { a[i] = lo[i]; a[8 + i] = hi[i]; }
    return a;
}
__device__ __forceinline__ v16h load_b_tile(const _Float16* __restrict__ W, int ld,
                                            int n0, int k0, int lane) {
    int row = n0 + (lane & 15);
    int kh  = k0 + ((lane >> 4) << 4);
    return *(const v16h*)(W + (size_t)row * ld + kh);
}

// ------------------- one-time conversion / init kernels -------------------
__global__ void cvt_w1_kernel(const float* __restrict__ wih, const float* __restrict__ whh,
                              _Float16* __restrict__ W1) {
    int n = blockIdx.x;                      // 2048 rows
    for (int k = threadIdx.x; k < K1D; k += blockDim.x) {
        float v = (k < EE + KVD) ? wih[(size_t)n * (EE + KVD) + k]
                                 : whh[(size_t)n * DHD + (k - (EE + KVD))];
        W1[(size_t)n * K1D + k] = (_Float16)v;
    }
}
__global__ void cvt_w2_kernel(const float* __restrict__ wih, const float* __restrict__ whh,
                              _Float16* __restrict__ W2) {
    int n = blockIdx.x;                      // 512 rows
    for (int k = threadIdx.x; k < K2D; k += blockDim.x) {
        float v = (k < DHD) ? wih[(size_t)n * DHD + k]
                            : whh[(size_t)n * KVD + (k - DHD)];
        W2[(size_t)n * K2D + k] = (_Float16)v;
    }
}
__global__ void cvt_emb_kernel(const float* __restrict__ e, _Float16* __restrict__ eh) {
    int i = blockIdx.x * blockDim.x + threadIdx.x;
    if (i < VV * EE) eh[i] = (_Float16)e[i];
}
__global__ void init_kernel(float* __restrict__ c1, float* __restrict__ c2,
                            _Float16* __restrict__ X2, const float* __restrict__ val,
                            _Float16* __restrict__ X) {
    int b = blockIdx.x, tid = threadIdx.x;   // 256 threads
    for (int j = tid; j < DHD; j += 256) {
        c1[(size_t)b * DHD + j] = 0.0f;
        X2[(size_t)b * K2D + j] = (_Float16)0.0f;          // h1 slot (t=0)
    }
    for (int j = tid; j < KVD; j += 256) {
        c2[(size_t)b * KVD + j] = 0.0f;
        X2[(size_t)b * K2D + DHD + j] = (_Float16)0.0f;    // h2 slot (t=0)
        float cv = val[(size_t)b * TT * KVD + j];          // ctx0 = value[:,0,:]
        X[(size_t)b * K1D + EE + j] = (_Float16)cv;
    }
}

// Writes token embedding for step t into X[:,0:256] and copies h1(prev) from X2 into X[:,384:896].
__global__ void embed_kernel(_Float16* __restrict__ X, const _Float16* __restrict__ emb_h,
                             const _Float16* __restrict__ X2, const int* __restrict__ y, int t) {
    int b = blockIdx.x, tid = threadIdx.x;   // 256 threads
    int tok = (t == 0) ? 0 : y[(size_t)b * LL + (t - 1)];
    X[(size_t)b * K1D + tid] = emb_h[(size_t)tok * EE + tid];
    X[(size_t)b * K1D + (EE + KVD) + tid]        = X2[(size_t)b * K2D + tid];
    X[(size_t)b * K1D + (EE + KVD) + 256 + tid]  = X2[(size_t)b * K2D + 256 + tid];
}

// ------------------- per-step kernels -------------------
// LSTM1: gates = X(256x896) @ W1cat^T(896x2048); fused gate nonlinearity + cell update.
__global__ void lstm1_kernel(const _Float16* __restrict__ X, const _Float16* __restrict__ W1,
                             const float* __restrict__ bih, const float* __restrict__ bhh,
                             float* __restrict__ c1, _Float16* __restrict__ X2) {
    int lane = threadIdx.x & 31;
    int wave = threadIdx.x >> 5;
    int m0 = blockIdx.x * 16;
    int n0 = blockIdx.y * 128 + wave * 16;

    v8f z = {};
    v8f acc[4] = { z, z, z, z };             // i, f, g, o
    for (int k0 = 0; k0 < K1D; k0 += 32) {
        v16h a = load_a_tile(X, K1D, m0, k0, lane);
#pragma unroll
        for (int g = 0; g < 4; ++g) {
            v16h bm = load_b_tile(W1, K1D, g * DHD + n0, k0, lane);
            acc[g] = __builtin_amdgcn_wmma_f32_16x16x32_f16(
                false, a, false, bm, (short)0, acc[g], false, false);
        }
    }
    int n = n0 + (lane & 15);
    float bi = bih[n]           + bhh[n];
    float bf = bih[DHD + n]     + bhh[DHD + n];
    float bg = bih[2 * DHD + n] + bhh[2 * DHD + n];
    float bo = bih[3 * DHD + n] + bhh[3 * DHD + n];
    int rbase = m0 + ((lane >> 4) << 3);
#pragma unroll
    for (int m = 0; m < 8; ++m) {
        int row = rbase + m;
        float gi = sigmoidf_(acc[0][m] + bi);
        float gf = sigmoidf_(acc[1][m] + bf);
        float gg = tanhf    (acc[2][m] + bg);
        float go = sigmoidf_(acc[3][m] + bo);
        float c  = gf * c1[(size_t)row * DHD + n] + gi * gg;
        c1[(size_t)row * DHD + n] = c;
        float h  = go * tanhf(c);
        X2[(size_t)row * K2D + n] = (_Float16)h;   // h1 -> lstm2 input / next-step X
    }
}

// LSTM2: gates = X2(256x640) @ W2cat^T(640x512); barrier before overwriting h2 slot.
__global__ void lstm2_kernel(_Float16* __restrict__ X2, const _Float16* __restrict__ W2,
                             const float* __restrict__ bih, const float* __restrict__ bhh,
                             float* __restrict__ c2, float* __restrict__ q) {
    int lane = threadIdx.x & 31;
    int wave = threadIdx.x >> 5;
    int m0 = blockIdx.x * 16;
    int n0 = wave * 16;

    v8f z = {};
    v8f acc[4] = { z, z, z, z };
    for (int k0 = 0; k0 < K2D; k0 += 32) {
        v16h a = load_a_tile(X2, K2D, m0, k0, lane);
#pragma unroll
        for (int g = 0; g < 4; ++g) {
            v16h bm = load_b_tile(W2, K2D, g * KVD + n0, k0, lane);
            acc[g] = __builtin_amdgcn_wmma_f32_16x16x32_f16(
                false, a, false, bm, (short)0, acc[g], false, false);
        }
    }
    __syncthreads();
    int n = n0 + (lane & 15);
    float bi = bih[n]           + bhh[n];
    float bf = bih[KVD + n]     + bhh[KVD + n];
    float bg = bih[2 * KVD + n] + bhh[2 * KVD + n];
    float bo = bih[3 * KVD + n] + bhh[3 * KVD + n];
    int rbase = m0 + ((lane >> 4) << 3);
#pragma unroll
    for (int m = 0; m < 8; ++m) {
        int row = rbase + m;
        float gi = sigmoidf_(acc[0][m] + bi);
        float gf = sigmoidf_(acc[1][m] + bf);
        float gg = tanhf    (acc[2][m] + bg);
        float go = sigmoidf_(acc[3][m] + bo);
        float c  = gf * c2[(size_t)row * KVD + n] + gi * gg;
        c2[(size_t)row * KVD + n] = c;
        float h  = go * tanhf(c);
        X2[(size_t)row * K2D + DHD + n] = (_Float16)h;  // new h2 for next step
        q[(size_t)row * KVD + n] = h;                   // f32 q for attention / projection
    }
}

// Fused: attention (energy/softmax/context) + output projection + next-step embed/h1-copy.
// One block per b, 256 threads. key_enc is read with default (RT) hints -> stays L2-resident;
// value is streamed with non-temporal hints so it can't evict key from the 192MB L2.
__global__ void attn_fused_kernel(const float* __restrict__ key, const float* __restrict__ val,
                                  const int* __restrict__ elen, const float* __restrict__ q,
                                  _Float16* __restrict__ X, const _Float16* __restrict__ X2,
                                  const _Float16* __restrict__ emb_h, const int* __restrict__ y,
                                  const float* __restrict__ embf, const float* __restrict__ ob,
                                  float* __restrict__ out, float* __restrict__ attn_row, int t) {
    __shared__ float qs[KVD];
    __shared__ float es[TT];
    __shared__ float red[256];
    __shared__ float ctmp[KVD];
    __shared__ float ctxs[KVD];
    int b = blockIdx.x, tid = threadIdx.x;
    if (tid < KVD) qs[tid] = q[(size_t)b * KVD + tid];
    __syncthreads();
    int len = elen[b];
    float lmax = -3.0e38f;
    for (int tt = tid; tt < TT; tt += 256) {
        const float4* kp = (const float4*)(key + ((size_t)b * TT + tt) * KVD);
        float acc = 0.0f;
#pragma unroll 8
        for (int k4 = 0; k4 < KVD / 4; ++k4) {
            float4 kk = kp[k4];
            acc += kk.x * qs[4 * k4] + kk.y * qs[4 * k4 + 1] +
                   kk.z * qs[4 * k4 + 2] + kk.w * qs[4 * k4 + 3];
        }
        acc = (tt < len) ? acc : -1.0e9f;
        es[tt] = acc;
        lmax = fmaxf(lmax, acc);
    }
    red[tid] = lmax; __syncthreads();
    for (int s = 128; s > 0; s >>= 1) {
        if (tid < s) red[tid] = fmaxf(red[tid], red[tid + s]);
        __syncthreads();
    }
    float mx = red[0]; __syncthreads();
    float lsum = 0.0f;
    for (int tt = tid; tt < TT; tt += 256) {
        float p = __expf(es[tt] - mx);
        es[tt] = p;
        lsum += p;
    }
    red[tid] = lsum; __syncthreads();
    for (int s = 128; s > 0; s >>= 1) {
        if (tid < s) red[tid] += red[tid + s];
        __syncthreads();
    }
    float inv = 1.0f / red[0];
    __syncthreads();
    // ctx[b,k] = sum_t p[t] * value[b,t,k]; t split by parity across the two half-blocks.
    // value is streamed non-temporally (bypass-ish L2 retention).
    int k = tid & (KVD - 1);
    int half = tid >> 7;
    float acc = 0.0f;
    {
        const fv4* vp = (const fv4*)(val + (size_t)b * TT * KVD);
        int k4 = k >> 2, kc = k & 3;
        for (int tt = half; tt < TT; tt += 2) {
            fv4 vv = __builtin_nontemporal_load(vp + (size_t)tt * (KVD / 4) + k4);
            acc += es[tt] * vv[kc];
        }
    }
    if (half) ctmp[k] = acc;
    __syncthreads();
    if (tid < KVD) {
        float c = (acc + ctmp[k]) * inv;
        ctxs[tid] = c;
        X[(size_t)b * K1D + EE + tid] = (_Float16)c;   // ctx slot for next step's lstm1
    }
    if (b == 0)
        for (int tt = tid; tt < TT; tt += 256) attn_row[tt] = es[tt] * inv;
    // next-step input prep: token embedding for step t+1 and h1 copy (lstm1(t) already done)
    {
        int tok = y[(size_t)b * LL + t];               // step t+1 consumes y[:, t]
        X[(size_t)b * K1D + tid] = emb_h[(size_t)tok * EE + tid];
        X[(size_t)b * K1D + (EE + KVD) + tid]       = X2[(size_t)b * K2D + tid];
        X[(size_t)b * K1D + (EE + KVD) + 256 + tid] = X2[(size_t)b * K2D + 256 + tid];
    }
    __syncthreads();
    // output projection: pred[b,v] = [q|ctx] . embedding[v,:] + out_bias[v]
    if (tid < VV) {
        float acc2 = ob[tid];
        const float* er = embf + (size_t)tid * EE;
#pragma unroll 4
        for (int kk = 0; kk < KVD; ++kk) acc2 += qs[kk] * er[kk];
#pragma unroll 4
        for (int kk = 0; kk < KVD; ++kk) acc2 += ctxs[kk] * er[KVD + kk];
        out[((size_t)b * LL + t) * VV + tid] = acc2;
    }
}

extern "C" void kernel_launch(void* const* d_in, const int* in_sizes, int n_in,
                              void* d_out, int out_size, void* d_ws, size_t ws_size,
                              hipStream_t stream) {
    const float* key_enc = (const float*)d_in[0];
    const float* value   = (const float*)d_in[1];
    const int*   elen    = (const int*)d_in[2];
    const int*   y       = (const int*)d_in[3];
    const float* emb     = (const float*)d_in[4];
    const float* w_ih1   = (const float*)d_in[5];
    const float* w_hh1   = (const float*)d_in[6];
    const float* b_ih1   = (const float*)d_in[7];
    const float* b_hh1   = (const float*)d_in[8];
    const float* w_ih2   = (const float*)d_in[9];
    const float* w_hh2   = (const float*)d_in[10];
    const float* b_ih2   = (const float*)d_in[11];
    const float* b_hh2   = (const float*)d_in[12];
    const float* obias   = (const float*)d_in[13];
    float* out = (float*)d_out;

    // carve workspace (d_ws is 256B-aligned)
    char* ws = (char*)d_ws;
    size_t off = 0;
    auto carve = [&](size_t bytes) -> void* {
        void* p = ws + off;
        off = (off + bytes + 255) & ~(size_t)255;
        return p;
    };
    _Float16* W1    = (_Float16*)carve((size_t)4 * DHD * K1D * 2);  // 2048 x 896 f16
    _Float16* W2    = (_Float16*)carve((size_t)4 * KVD * K2D * 2);  // 512 x 640 f16
    _Float16* emb_h = (_Float16*)carve((size_t)VV * EE * 2);
    _Float16* X     = (_Float16*)carve((size_t)BB * K1D * 2);       // [emb|ctx|h1] f16
    _Float16* X2    = (_Float16*)carve((size_t)BB * K2D * 2);       // [h1|h2] f16
    float*    c1    = (float*)carve((size_t)BB * DHD * 4);
    float*    c2    = (float*)carve((size_t)BB * KVD * 4);
    float*    qf    = (float*)carve((size_t)BB * KVD * 4);

    // one-time prep (runs every call; deterministic)
    cvt_w1_kernel<<<4 * DHD, 256, 0, stream>>>(w_ih1, w_hh1, W1);
    cvt_w2_kernel<<<4 * KVD, 256, 0, stream>>>(w_ih2, w_hh2, W2);
    cvt_emb_kernel<<<(VV * EE + 255) / 256, 256, 0, stream>>>(emb, emb_h);
    init_kernel<<<BB, 256, 0, stream>>>(c1, c2, X2, value, X);
    embed_kernel<<<BB, 256, 0, stream>>>(X, emb_h, X2, y, 0);  // SOS + zero h1 copy

    float* attn_base = out + (size_t)BB * LL * VV;
    for (int t = 0; t < LL; ++t) {
        lstm1_kernel<<<dim3(BB / 16, 4), 256, 0, stream>>>(X, W1, b_ih1, b_hh1, c1, X2);
        lstm2_kernel<<<BB / 16, 256, 0, stream>>>(X2, W2, b_ih2, b_hh2, c2, qf);
        attn_fused_kernel<<<BB, 256, 0, stream>>>(key_enc, value, elen, qf, X, X2,
                                                  emb_h, y, emb, obias, out,
                                                  attn_base + (size_t)t * TT, t);
    }
}
